// VINNetwork_54769422959171
// MI455X (gfx1250) — compile-verified, matrix operand order checked by reading, not run on hardware
//
#include <hip/hip_runtime.h>
#include <stdint.h>

#define CELLS 4096
#define K_ITERS 10

typedef __attribute__((ext_vector_type(16))) __bf16        v16bf;
typedef __attribute__((ext_vector_type(8)))  float         v8f;
typedef __attribute__((ext_vector_type(4)))  unsigned int  u32x4;

__device__ __forceinline__ unsigned short f2bf(float x) {
    unsigned int u = __float_as_uint(x);
    u += 0x7FFFu + ((u >> 16) & 1u);          // round-to-nearest-even
    return (unsigned short)(u >> 16);
}

union TileU { u32x4 q[2]; v16bf v; };

// Load one 16x32 bf16 fragment row for WMMA A/B from a [rows][kpad] bf16 panel.
// ISA 16-bit A layout: lane half h covers K = {kbase+h*8 .. +7} then {kbase+16+h*8 .. +7}
// -> two contiguous, 16B-aligned runs per lane.
__device__ __forceinline__ v16bf load_tile16(const unsigned short* base, int row,
                                             int kpad, int kbase, int half) {
    TileU t;
    const unsigned short* rp = base + (size_t)row * kpad + kbase + half * 8;
    t.q[0] = *reinterpret_cast<const u32x4*>(rp);
    t.q[1] = *reinterpret_cast<const u32x4*>(rp + 16);
    return t.v;
}

// ---------------------------------------------------------------------------
// Weight repack: f32 [K][N] -> bf16 [N][Kpad] (K-contiguous, zero padded)
// W1 [36][256] -> W1T [256][64]; W2 [256][256] -> W2T [256][256];
// W_logits [256][5] -> WLT [16][256] (rows n>=5 zeroed).
// ---------------------------------------------------------------------------
__global__ __launch_bounds__(256) void pack_weights(
    const float* __restrict__ W1, const float* __restrict__ W2,
    const float* __restrict__ WL,
    unsigned short* __restrict__ W1T, unsigned short* __restrict__ W2T,
    unsigned short* __restrict__ WLT) {
    int i = blockIdx.x * 256 + threadIdx.x;
    if (i < 16384) {                       // W1T: 256*64
        int n = i >> 6, k = i & 63;
        W1T[i] = (k < 36) ? f2bf(W1[k * 256 + n]) : (unsigned short)0;
    } else if (i < 81920) {                // W2T: 256*256
        int j = i - 16384;
        int n = j >> 8, k = j & 255;
        W2T[j] = f2bf(W2[k * 256 + n]);
    } else if (i < 86016) {                // WLT: 16*256
        int j = i - 81920;
        int n = j >> 8, k = j & 255;
        WLT[j] = (n < 5) ? f2bf(WL[k * 5 + n]) : (unsigned short)0;
    }
}

// ---------------------------------------------------------------------------
// Per-batch-element: phi -> K=10 value-iteration sweeps entirely in LDS ->
// 36 agent-window features (bf16, zero-padded to K=64).
// One workgroup (256 threads, 8 waves) per batch element; 64 KB LDS.
// ---------------------------------------------------------------------------
__global__ __launch_bounds__(256) void vin_feats_kernel(
    const float* __restrict__ obs, const float* __restrict__ Wphi,
    const float* __restrict__ bphi, unsigned short* __restrict__ feats) {
    __shared__ float sp[CELLS];
    __shared__ float srin[CELLS];
    __shared__ float srout[CELLS];
    __shared__ float sv[CELLS];

    const int tid = threadIdx.x;
    const int b   = blockIdx.x;
    const float* ob = obs + (size_t)b * CELLS * 3;

    // ---- agent = argmax(one-hot channel 1) = min set index ----
    if (tid == 0) ((int*)sv)[0] = 0x7FFFFFFF;
    __syncthreads();
    int cand = 0x7FFFFFFF;
    for (int c = tid; c < CELLS; c += 256)
        if (ob[c * 3 + 1] > 0.5f && c < cand) cand = c;
    if (cand != 0x7FFFFFFF) atomicMin((int*)sv, cand);
    __syncthreads();
    const int agent = ((int*)sv)[0];
    __syncthreads();

    // ---- phi: 3->3 linear + relu, keep p/rin/rout in LDS; v0 = 0 ----
    const float w00 = Wphi[0], w01 = Wphi[1], w02 = Wphi[2];
    const float w10 = Wphi[3], w11 = Wphi[4], w12 = Wphi[5];
    const float w20 = Wphi[6], w21 = Wphi[7], w22 = Wphi[8];
    const float bb0 = bphi[0], bb1 = bphi[1], bb2 = bphi[2];
    for (int c = tid; c < CELLS; c += 256) {
        float x0 = ob[c * 3 + 0], x1 = ob[c * 3 + 1], x2 = ob[c * 3 + 2];
        sp[c]    = fmaxf(0.f, fmaf(x0, w00, fmaf(x1, w10, fmaf(x2, w20, bb0))));
        srin[c]  = fmaxf(0.f, fmaf(x0, w01, fmaf(x1, w11, fmaf(x2, w21, bb1))));
        srout[c] = fmaxf(0.f, fmaf(x0, w02, fmaf(x1, w12, fmaf(x2, w22, bb2))));
        sv[c]    = 0.f;
    }
    __syncthreads();

    // ---- K synchronous sweeps; register-staged in-place update ----
    for (int it = 0; it < K_ITERS; ++it) {
        float nv[16];
#pragma unroll
        for (int t = 0; t < 16; ++t) {
            int c = tid + t * 256;
            int i = c >> 6, j = c & 63;
            float p = sp[c], ro = srout[c];
            float best = sv[c];
            if (i > 0)  best = fmaxf(best, fmaf(p, sv[c - 64], srin[c - 64]) - ro);
            if (i < 63) best = fmaxf(best, fmaf(p, sv[c + 64], srin[c + 64]) - ro);
            if (j > 0)  best = fmaxf(best, fmaf(p, sv[c - 1],  srin[c - 1])  - ro);
            if (j < 63) best = fmaxf(best, fmaf(p, sv[c + 1],  srin[c + 1])  - ro);
            nv[t] = best;
        }
        __syncthreads();
#pragma unroll
        for (int t = 0; t < 16; ++t) sv[tid + t * 256] = nv[t];
        __syncthreads();
    }

    // ---- 36 features (3x3 windows of 1-walls, agent, goal, v), pad to 64 ----
    if (tid < 64) {
        float val = 0.f;
        if (tid < 36) {
            int g = tid / 9, idx = tid % 9;
            int r = (agent >> 6) - 1 + idx / 3;
            int c = (agent & 63) - 1 + idx % 3;
            if (r >= 0 && r < 64 && c >= 0 && c < 64) {
                int cc = r * 64 + c;
                if      (g == 0) val = 1.f - ob[cc * 3 + 0];
                else if (g == 1) val = ob[cc * 3 + 1];
                else if (g == 2) val = ob[cc * 3 + 2];
                else             val = sv[cc];
            }
        }
        feats[(size_t)b * 64 + tid] = f2bf(val);
    }
}

// ---------------------------------------------------------------------------
// C[M,N] = bf16(A[M,Kpad] @ Bt[N,Kpad]^T + bias); one wave per 16x16 tile.
// ---------------------------------------------------------------------------
__global__ __launch_bounds__(256) void gemm_bf16_bias(
    const unsigned short* __restrict__ A, const unsigned short* __restrict__ Bt,
    const float* __restrict__ bias, unsigned short* __restrict__ outb,
    int M, int N, int Kpad) {
    int wave = (blockIdx.x * blockDim.x + threadIdx.x) >> 5;
    int ntn = N >> 4;
    if (wave >= (M >> 4) * ntn) return;
    int tm = wave / ntn, tn = wave % ntn;
    int lane = threadIdx.x & 31;
    int half = lane >> 4, lm = lane & 15;

    v8f acc = {};
    for (int ks = 0; ks < Kpad; ks += 32) {
        v16bf a = load_tile16(A,  tm * 16 + lm, Kpad, ks, half);
        v16bf w = load_tile16(Bt, tn * 16 + lm, Kpad, ks, half);
        acc = __builtin_amdgcn_wmma_f32_16x16x32_bf16(
            false, a, false, w, (short)0, acc, false, false);
    }
    int gn = tn * 16 + lm;
    float bv = bias[gn];
#pragma unroll
    for (int r = 0; r < 8; ++r) {
        int gm = tm * 16 + half * 8 + r;       // D layout: vgpr r -> M=r (+8 for hi half)
        outb[(size_t)gm * N + gn] = f2bf(acc[r] + bv);
    }
}

// Final layer: [M,Kpad] @ WLT[16,Kpad]^T, write f32 logits cols 0..4.
__global__ __launch_bounds__(256) void gemm_bf16_logits(
    const unsigned short* __restrict__ A, const unsigned short* __restrict__ Bt,
    const float* __restrict__ bias, float* __restrict__ out, int M, int Kpad) {
    int wave = (blockIdx.x * blockDim.x + threadIdx.x) >> 5;
    if (wave >= (M >> 4)) return;
    int tm = wave;
    int lane = threadIdx.x & 31;
    int half = lane >> 4, lm = lane & 15;

    v8f acc = {};
    for (int ks = 0; ks < Kpad; ks += 32) {
        v16bf a = load_tile16(A,  tm * 16 + lm, Kpad, ks, half);
        v16bf w = load_tile16(Bt, lm,           Kpad, ks, half);
        acc = __builtin_amdgcn_wmma_f32_16x16x32_bf16(
            false, a, false, w, (short)0, acc, false, false);
    }
    if (lm < 5) {                               // divergence only after all WMMAs
        float bv = bias[lm];
#pragma unroll
        for (int r = 0; r < 8; ++r) {
            int gm = tm * 16 + half * 8 + r;
            out[(size_t)gm * 5 + lm] = acc[r] + bv;
        }
    }
}

extern "C" void kernel_launch(void* const* d_in, const int* in_sizes, int n_in,
                              void* d_out, int out_size, void* d_ws, size_t ws_size,
                              hipStream_t stream) {
    const float* obs  = (const float*)d_in[0];
    const float* Wphi = (const float*)d_in[1];
    const float* bphi = (const float*)d_in[2];
    const float* W1   = (const float*)d_in[3];
    const float* b1   = (const float*)d_in[4];
    const float* W2   = (const float*)d_in[5];
    const float* b2   = (const float*)d_in[6];
    const float* WL   = (const float*)d_in[7];
    const float* bL   = (const float*)d_in[8];
    float* out = (float*)d_out;
    char*  ws  = (char*)d_ws;

    unsigned short* W1T   = (unsigned short*)(ws + 0);        //  32768 B [256][64]
    unsigned short* W2T   = (unsigned short*)(ws + 32768);    // 131072 B [256][256]
    unsigned short* WLT   = (unsigned short*)(ws + 163840);   //   8192 B [16][256]
    unsigned short* feats = (unsigned short*)(ws + 172032);   // 131072 B [1024][64]
    unsigned short* h1    = (unsigned short*)(ws + 303104);   // 524288 B [1024][256]
    unsigned short* h2    = (unsigned short*)(ws + 827392);   // 524288 B [1024][256]

    pack_weights<<<336, 256, 0, stream>>>(W1, W2, WL, W1T, W2T, WLT);
    vin_feats_kernel<<<1024, 256, 0, stream>>>(obs, Wphi, bphi, feats);
    // feats[1024,64] @ W1T -> h1[1024,256]   (64*16 tiles / 8 waves = 128 blocks)
    gemm_bf16_bias<<<128, 256, 0, stream>>>(feats, W1T, b1, h1, 1024, 256, 64);
    // h1 @ W2T -> h2[1024,256]
    gemm_bf16_bias<<<128, 256, 0, stream>>>(h1, W2T, b2, h2, 1024, 256, 256);
    // h2 @ WLT -> logits[1024,5]             (64 tiles / 8 waves = 8 blocks)
    gemm_bf16_logits<<<8, 256, 0, stream>>>(h2, WLT, bL, out, 1024, 256);
}
